// GraphNetAutoCenter_67482526155001
// MI455X (gfx1250) — matrix operand order repaired and checked by reading, not compiled, not run on hardware
//
#include <hip/hip_runtime.h>

#define N_V 100000
#define D_F 128
#define N_E 1600000

typedef __attribute__((ext_vector_type(16))) __bf16        v16bf;
typedef __attribute__((ext_vector_type(8)))  float         v8f;
typedef __attribute__((ext_vector_type(4)))  unsigned int  u32x4;

union Frag16 { v16bf v; u32x4 q[2]; unsigned short h[16]; };

// hardware round-to-nearest-even f32 -> bf16
__device__ __forceinline__ unsigned short bf16b(float f) {
  union { __bf16 b; unsigned short u; } c;
  c.b = (__bf16)f;
  return c.u;
}

__device__ __forceinline__ v8f v8f_zero() {
  v8f z = {0.f, 0.f, 0.f, 0.f, 0.f, 0.f, 0.f, 0.f};
  return z;
}

// wave-local LDS ordering (DS ops are in-order within a wave; this orders the compiler)
__device__ __forceinline__ void wave_lds_fence() {
  __builtin_amdgcn_fence(__ATOMIC_ACQ_REL, "wavefront");
}

// Load a B fragment from W^T storage: 8 bf16 at col*128 + kbase and 8 at +16.
__device__ __forceinline__ v16bf load_bfragT(const unsigned short* __restrict__ wT,
                                             int col, int kbase) {
  Frag16 b;
  const unsigned short* p = wT + col * 128 + kbase;
  b.q[0] = *(const u32x4*)p;
  b.q[1] = *(const u32x4*)(p + 16);
  return b.v;
}

// ---------------- prep kernels ----------------

__global__ void k_cvt_bf16(const float* __restrict__ s, unsigned short* __restrict__ d, int n) {
  int i = blockIdx.x * blockDim.x + threadIdx.x;
  if (i < n) d[i] = bf16b(s[i]);
}

// dst[col*K + k] = bf16(src[k*N + col]); src is [K][N] row-major, dst is W^T [N][K].
__global__ void k_cvt_bf16_t(const float* __restrict__ src, unsigned short* __restrict__ dst,
                             int K, int N) {
  int i = blockIdx.x * blockDim.x + threadIdx.x;
  if (i < K * N) {
    int col = i / K;
    int k = i - col * K;
    dst[i] = bf16b(src[k * N + col]);
  }
}

__global__ void k_zero_f32(float* __restrict__ p, int n) {
  int i = blockIdx.x * blockDim.x + threadIdx.x;
  if (i < n) p[i] = 0.f;
}

// ---------------- offset MLP: centered = coords + relu(vf@Wo1+bo1)@Wo2+bo2 ----------------

__global__ __launch_bounds__(128) void k_offset(
    const unsigned short* __restrict__ vf_bf,    // [N][128] bf16
    const float* __restrict__ coords,            // [N][3]
    const unsigned short* __restrict__ wo1T_bf,  // [64][128] bf16 (W^T)
    const float* __restrict__ bo1,               // [64]
    const float* __restrict__ Wo2,               // [64][3]
    const float* __restrict__ bo2,               // [3]
    float* __restrict__ centered)                // [N][3]
{
  __shared__ unsigned short sWT[64 * 128];
  __shared__ float sT[4][16 * 64];

  const int tid = threadIdx.x;
  {
    const u32x4* s = (const u32x4*)wo1T_bf;
    u32x4* d = (u32x4*)sWT;
    for (int i = tid; i < 64 * 128 / 8; i += 128) d[i] = s[i];
  }
  __syncthreads();

  const int wave = tid >> 5, lane = tid & 31, hh = lane >> 4, l16 = lane & 15;
  const int wid = blockIdx.x * 4 + wave;

  float bo1f[4];
#pragma unroll
  for (int n = 0; n < 4; ++n) bo1f[n] = bo1[16 * n + l16];

  for (int t = 0; t < 8; ++t) {
    const int tile = wid * 8 + t;
    if (tile >= N_V / 16) break;
    const int v0 = tile * 16;
    const int vr = v0 + l16;

    v8f acc[4];
#pragma unroll
    for (int n = 0; n < 4; ++n) acc[n] = v8f_zero();

#pragma unroll
    for (int k = 0; k < 4; ++k) {
      Frag16 a;
      const unsigned short* p = vf_bf + (size_t)vr * 128 + 32 * k + 8 * hh;
      a.q[0] = *(const u32x4*)p;
      a.q[1] = *(const u32x4*)(p + 16);
      const int kb = 32 * k + 8 * hh;
#pragma unroll
      for (int n = 0; n < 4; ++n) {
        v16bf b = load_bfragT(sWT, 16 * n + l16, kb);
        acc[n] = __builtin_amdgcn_wmma_f32_16x16x32_bf16(false, a.v, false, b,
                                                         (short)0, acc[n], false, false);
      }
    }

#pragma unroll
    for (int n = 0; n < 4; ++n) {
      const int col = 16 * n + l16;
#pragma unroll
      for (int r = 0; r < 8; ++r)
        sT[wave][(r + 8 * hh) * 64 + col] = fmaxf(acc[n][r] + bo1f[n], 0.f);
    }
    wave_lds_fence();

    if (hh == 0) {
      const int v = v0 + l16;
      float o0 = bo2[0], o1 = bo2[1], o2 = bo2[2];
#pragma unroll 8
      for (int j = 0; j < 64; ++j) {
        const float x = sT[wave][l16 * 64 + j];
        o0 += x * Wo2[j * 3 + 0];
        o1 += x * Wo2[j * 3 + 1];
        o2 += x * Wo2[j * 3 + 2];
      }
      centered[3 * v + 0] = coords[3 * v + 0] + o0;
      centered[3 * v + 1] = coords[3 * v + 1] + o1;
      centered[3 * v + 2] = coords[3 * v + 2] + o2;
    }
    wave_lds_fence();
  }
}

// ---------------- edge MLP + segment-max scatter ----------------

__global__ __launch_bounds__(128) void k_edge(
    const unsigned short* __restrict__ vf_bf,    // [N][128] bf16
    const float* __restrict__ coords,            // [N][3]
    const float* __restrict__ centered,          // [N][3]
    const int* __restrict__ edges,               // [E][2]
    const unsigned short* __restrict__ we1T_bf,  // [128][128] bf16 (W^T rows 0..127)
    const float* __restrict__ We1,               // [131][128] f32 (rows 128..130 used)
    const float* __restrict__ be1,               // [128]
    const unsigned short* __restrict__ we2T_bf,  // [128][128] bf16 (W^T)
    const float* __restrict__ be2,               // [128]
    float* __restrict__ agg)                     // [N][128], >=0, int-atomicMax
{
  __shared__ unsigned short sW1T[128 * 128];
  __shared__ unsigned short sH[4][16 * 128];
  __shared__ float sCD[4][16][3];
  __shared__ int sSrc[4][16];
  __shared__ int sDst[4][16];

  const int tid = threadIdx.x;
  {
    const u32x4* s = (const u32x4*)we1T_bf;
    u32x4* d = (u32x4*)sW1T;
    for (int i = tid; i < 128 * 128 / 8; i += 128) d[i] = s[i];
  }
  __syncthreads();

  const int wave = tid >> 5, lane = tid & 31, hh = lane >> 4, l16 = lane & 15;
  const int wid = blockIdx.x * 4 + wave;

  // per-lane constants across all tiles
  float be1f[8], be2f[8], w0f[8], w1f[8], w2f[8];
#pragma unroll
  for (int n = 0; n < 8; ++n) {
    const int col = 16 * n + l16;
    be1f[n] = be1[col];
    be2f[n] = be2[col];
    w0f[n] = We1[128 * 128 + col];
    w1f[n] = We1[129 * 128 + col];
    w2f[n] = We1[130 * 128 + col];
  }

  for (int t = 0; t < 8; ++t) {
    const int e0 = (wid * 8 + t) * 16;

    if (hh == 0) {
      const int e = e0 + l16;
      const int s = edges[2 * e + 0];
      const int d = edges[2 * e + 1];
      sSrc[wave][l16] = s;
      sDst[wave][l16] = d;
      sCD[wave][l16][0] = coords[3 * s + 0] - centered[3 * d + 0];
      sCD[wave][l16][1] = coords[3 * s + 1] - centered[3 * d + 1];
      sCD[wave][l16][2] = coords[3 * s + 2] - centered[3 * d + 2];
    }
    wave_lds_fence();

    const int src = sSrc[wave][l16];

    v8f acc[8];
#pragma unroll
    for (int n = 0; n < 8; ++n) acc[n] = v8f_zero();

    // layer 1: feature part via WMMA (K = 0..127)
#pragma unroll
    for (int k = 0; k < 4; ++k) {
      Frag16 a;
      const unsigned short* p = vf_bf + (size_t)src * 128 + 32 * k + 8 * hh;
      a.q[0] = *(const u32x4*)p;
      a.q[1] = *(const u32x4*)(p + 16);
      const int kb = 32 * k + 8 * hh;
#pragma unroll
      for (int n = 0; n < 8; ++n) {
        v16bf b = load_bfragT(sW1T, 16 * n + l16, kb);
        acc[n] = __builtin_amdgcn_wmma_f32_16x16x32_bf16(false, a.v, false, b,
                                                         (short)0, acc[n], false, false);
      }
    }

    // coordinate rows (K = 128..130) + bias + relu -> bf16 into LDS
    float cda[8], cdb[8], cdc[8];
#pragma unroll
    for (int r = 0; r < 8; ++r) {
      cda[r] = sCD[wave][r + 8 * hh][0];
      cdb[r] = sCD[wave][r + 8 * hh][1];
      cdc[r] = sCD[wave][r + 8 * hh][2];
    }
#pragma unroll
    for (int n = 0; n < 8; ++n) {
      const int col = 16 * n + l16;
#pragma unroll
      for (int r = 0; r < 8; ++r) {
        float v = acc[n][r] + be1f[n] + cda[r] * w0f[n] + cdb[r] * w1f[n] + cdc[r] * w2f[n];
        sH[wave][(r + 8 * hh) * 128 + col] = bf16b(fmaxf(v, 0.f));
      }
    }
    wave_lds_fence();

    // layer 2: h @ We2 (K = 128), B from L2-resident transposed weights
#pragma unroll
    for (int n = 0; n < 8; ++n) acc[n] = v8f_zero();
#pragma unroll
    for (int k = 0; k < 4; ++k) {
      Frag16 a;
      const unsigned short* p = &sH[wave][l16 * 128 + 32 * k + 8 * hh];
      a.q[0] = *(const u32x4*)p;
      a.q[1] = *(const u32x4*)(p + 16);
      const int kb = 32 * k + 8 * hh;
#pragma unroll
      for (int n = 0; n < 8; ++n) {
        v16bf b = load_bfragT(we2T_bf, 16 * n + l16, kb);
        acc[n] = __builtin_amdgcn_wmma_f32_16x16x32_bf16(false, a.v, false, b,
                                                         (short)0, acc[n], false, false);
      }
    }

    int dstv[8];
#pragma unroll
    for (int r = 0; r < 8; ++r) dstv[r] = sDst[wave][r + 8 * hh];
#pragma unroll
    for (int n = 0; n < 8; ++n) {
      const int col = 16 * n + l16;
#pragma unroll
      for (int r = 0; r < 8; ++r) {
        const float v = fmaxf(acc[n][r] + be2f[n], 0.f);  // >= 0: int max == float max
        atomicMax((int*)&agg[(size_t)dstv[r] * 128 + col], __float_as_int(v));
      }
    }
    wave_lds_fence();
  }
}

// ---------------- update MLP: out = relu(agg@Wu1+bu1)@Wu2 + bu2 + vf ----------------

__global__ __launch_bounds__(128) void k_update(
    const float* __restrict__ agg,               // [N][128]
    const unsigned short* __restrict__ wu1T_bf,  // [128][128] bf16 (W^T)
    const float* __restrict__ bu1,               // [128]
    const unsigned short* __restrict__ wu2T_bf,  // [128][128] bf16 (W^T)
    const float* __restrict__ bu2,               // [128]
    const float* __restrict__ vf,                // [N][128]
    float* __restrict__ out)                     // [N][128]
{
  __shared__ unsigned short sWT[128 * 128];
  __shared__ unsigned short sH[4][16 * 128];

  const int tid = threadIdx.x;
  {
    const u32x4* s = (const u32x4*)wu1T_bf;
    u32x4* d = (u32x4*)sWT;
    for (int i = tid; i < 128 * 128 / 8; i += 128) d[i] = s[i];
  }
  __syncthreads();

  const int wave = tid >> 5, lane = tid & 31, hh = lane >> 4, l16 = lane & 15;
  const int wid = blockIdx.x * 4 + wave;

  float bu1f[8], bu2f[8];
#pragma unroll
  for (int n = 0; n < 8; ++n) {
    bu1f[n] = bu1[16 * n + l16];
    bu2f[n] = bu2[16 * n + l16];
  }

  for (int t = 0; t < 8; ++t) {
    const int tile = wid * 8 + t;
    if (tile >= N_V / 16) break;
    const int v0 = tile * 16;
    const int vr = v0 + l16;

    v8f acc[8];
#pragma unroll
    for (int n = 0; n < 8; ++n) acc[n] = v8f_zero();

#pragma unroll
    for (int k = 0; k < 4; ++k) {
      Frag16 a;
      const float* p = agg + (size_t)vr * 128 + 32 * k + 8 * hh;
#pragma unroll
      for (int i = 0; i < 8; ++i) a.h[i] = bf16b(p[i]);
#pragma unroll
      for (int i = 0; i < 8; ++i) a.h[8 + i] = bf16b(p[16 + i]);
      const int kb = 32 * k + 8 * hh;
#pragma unroll
      for (int n = 0; n < 8; ++n) {
        v16bf b = load_bfragT(sWT, 16 * n + l16, kb);
        acc[n] = __builtin_amdgcn_wmma_f32_16x16x32_bf16(false, a.v, false, b,
                                                         (short)0, acc[n], false, false);
      }
    }

#pragma unroll
    for (int n = 0; n < 8; ++n) {
      const int col = 16 * n + l16;
#pragma unroll
      for (int r = 0; r < 8; ++r)
        sH[wave][(r + 8 * hh) * 128 + col] = bf16b(fmaxf(acc[n][r] + bu1f[n], 0.f));
    }
    wave_lds_fence();

#pragma unroll
    for (int n = 0; n < 8; ++n) acc[n] = v8f_zero();
#pragma unroll
    for (int k = 0; k < 4; ++k) {
      Frag16 a;
      const unsigned short* p = &sH[wave][l16 * 128 + 32 * k + 8 * hh];
      a.q[0] = *(const u32x4*)p;
      a.q[1] = *(const u32x4*)(p + 16);
      const int kb = 32 * k + 8 * hh;
#pragma unroll
      for (int n = 0; n < 8; ++n) {
        v16bf b = load_bfragT(wu2T_bf, 16 * n + l16, kb);
        acc[n] = __builtin_amdgcn_wmma_f32_16x16x32_bf16(false, a.v, false, b,
                                                         (short)0, acc[n], false, false);
      }
    }

#pragma unroll
    for (int n = 0; n < 8; ++n) {
      const int col = 16 * n + l16;
#pragma unroll
      for (int r = 0; r < 8; ++r) {
        const int v = v0 + r + 8 * hh;
        out[(size_t)v * 128 + col] = acc[n][r] + bu2f[n] + vf[(size_t)v * 128 + col];
      }
    }
    wave_lds_fence();
  }
}

// ---------------- launcher ----------------

extern "C" void kernel_launch(void* const* d_in, const int* in_sizes, int n_in,
                              void* d_out, int out_size, void* d_ws, size_t ws_size,
                              hipStream_t stream) {
  const float* vf     = (const float*)d_in[0];
  const float* coords = (const float*)d_in[1];
  const int*   edges  = (const int*)d_in[2];
  const float* Wo1 = (const float*)d_in[3];
  const float* bo1 = (const float*)d_in[4];
  const float* Wo2 = (const float*)d_in[5];
  const float* bo2 = (const float*)d_in[6];
  const float* We1 = (const float*)d_in[7];
  const float* be1 = (const float*)d_in[8];
  const float* We2 = (const float*)d_in[9];
  const float* be2 = (const float*)d_in[10];
  const float* Wu1 = (const float*)d_in[11];
  const float* bu1 = (const float*)d_in[12];
  const float* Wu2 = (const float*)d_in[13];
  const float* bu2 = (const float*)d_in[14];
  float* out = (float*)d_out;

  char* ws = (char*)d_ws;
  size_t off = 0;
  auto alloc = [&](size_t bytes) -> char* {
    char* p = ws + off;
    off = (off + bytes + 255) & ~(size_t)255;
    return p;
  };
  unsigned short* vf_bf   = (unsigned short*)alloc((size_t)N_V * 128 * 2);
  unsigned short* wo1T_bf = (unsigned short*)alloc(64 * 128 * 2);
  unsigned short* we1T_bf = (unsigned short*)alloc(128 * 128 * 2);
  unsigned short* we2T_bf = (unsigned short*)alloc(128 * 128 * 2);
  unsigned short* wu1T_bf = (unsigned short*)alloc(128 * 128 * 2);
  unsigned short* wu2T_bf = (unsigned short*)alloc(128 * 128 * 2);
  float* centered = (float*)alloc((size_t)N_V * 3 * 4);
  float* agg      = (float*)alloc((size_t)N_V * 128 * 4);

  k_cvt_bf16<<<(N_V * 128 + 255) / 256, 256, 0, stream>>>(vf, vf_bf, N_V * 128);

  k_cvt_bf16_t<<<(128 * 64 + 255) / 256, 256, 0, stream>>>(Wo1, wo1T_bf, 128, 64);
  k_cvt_bf16_t<<<(128 * 128 + 255) / 256, 256, 0, stream>>>(We1, we1T_bf, 128, 128);
  k_cvt_bf16_t<<<(128 * 128 + 255) / 256, 256, 0, stream>>>(We2, we2T_bf, 128, 128);
  k_cvt_bf16_t<<<(128 * 128 + 255) / 256, 256, 0, stream>>>(Wu1, wu1T_bf, 128, 128);
  k_cvt_bf16_t<<<(128 * 128 + 255) / 256, 256, 0, stream>>>(Wu2, wu2T_bf, 128, 128);

  k_zero_f32<<<(N_V * 128 + 255) / 256, 256, 0, stream>>>(agg, N_V * 128);

  // vertex kernels: 6250 tiles, 8 tiles per wave, 4 waves per block
  const int vblocks = (N_V / 16 + 31) / 32;     // 196
  // edge kernel: 100000 tiles, 8 per wave, 4 waves per block -> exact
  const int eblocks = N_E / (16 * 8 * 4);       // 3125

  k_offset<<<vblocks, 128, 0, stream>>>(vf_bf, coords, wo1T_bf, bo1, Wo2, bo2, centered);
  k_edge<<<eblocks, 128, 0, stream>>>(vf_bf, coords, centered, edges,
                                      we1T_bf, We1, be1, we2T_bf, be2, agg);
  k_update<<<vblocks, 128, 0, stream>>>(agg, wu1T_bf, bu1, wu2T_bf, bu2, vf, out);
}
